// VQVAE_17703855194499
// MI455X (gfx1250) — matrix operand order repaired and checked by reading, not compiled, not run on hardware
//
#include <hip/hip_runtime.h>

typedef __attribute__((ext_vector_type(16))) _Float16 v16h;
typedef __attribute__((ext_vector_type(8)))  float    v8f;

struct Taps { int n; int dy[16]; int dx[16]; };
struct KMap { int idx[16]; };

#define GEMM_BM 64
#define GEMM_BN 128
#define GEMM_TK 32
#define LDA_H 40   // padded row stride (halves) for A tile in LDS
#define LDB_H 40   // padded row stride (halves) for B tile in LDS (rows = n, cols = k)

__device__ __forceinline__ unsigned pack_h2(float a, float b) {
    unsigned short lo = __builtin_bit_cast(unsigned short, (_Float16)a);
    unsigned short hi = __builtin_bit_cast(unsigned short, (_Float16)b);
    return (unsigned)lo | ((unsigned)hi << 16);
}

// ---------------------------------------------------------------------------
// Generic implicit-GEMM convolution using v_wmma_f32_16x16x32_f16.
//  D[co, pos] = sum_k A[co,k] * patch(k,pos),  k = tap*Cin + ci
//  Output written at (oy0 + ty*oys, ox0 + tx*oxs) — handles pad-border
//  interiors (eo/do) and stride-2 scatter (ConvT parity classes).
// ---------------------------------------------------------------------------
__launch_bounds__(128)
__global__ void conv_gemm_wmma(const _Float16* __restrict__ A,
                               const float* __restrict__ X,
                               const float* __restrict__ bias,
                               float* __restrict__ Y,
                               Taps taps, int Cin, int IH, int IW, int stride,
                               int relu_in, int Ktot, int Kpad,
                               int OHt, int OWt, int oy0, int ox0, int oys, int oxs,
                               int OHf, int OWf, int Cout)
{
    __shared__ _Float16 As[GEMM_BM * LDA_H];
    __shared__ _Float16 Bs[GEMM_BN * LDB_H];
    unsigned* Bs32 = (unsigned*)Bs;    // LDB_H even, kk even -> aligned b32 view

    const int tid   = threadIdx.x;
    const int lane  = tid & 31;
    const int wave  = tid >> 5;
    const int half_ = lane >> 4;   // 0: lanes 0-15, 1: lanes 16-31
    const int l16   = lane & 15;
    const int img   = blockIdx.z;
    const int mBase = blockIdx.y * GEMM_BM;
    const int Npos  = OHt * OWt;
    const int posBase = blockIdx.x * GEMM_BN;
    const int wm = (wave >> 1) * 32;   // wave M offset inside 64
    const int wn = (wave & 1) * 64;    // wave N offset inside 128

    const float* Xi = X + (long)img * Cin * IH * IW;
    const long IHW  = (long)IH * IW;
    const bool aligned = (Cin % GEMM_TK) == 0;   // whole chunk within one tap

    v8f acc[2][4];
#pragma unroll
    for (int a = 0; a < 2; ++a)
#pragma unroll
        for (int b = 0; b < 4; ++b)
#pragma unroll
            for (int r = 0; r < 8; ++r) acc[a][b][r] = 0.0f;

    // B staging: this thread owns one N column
    const int  nB = tid;
    const int  posB = posBase + nB;
    const bool validB = posB < Npos;
    const int  tyB = validB ? (posB / OWt) : 0;
    const int  txB = validB ? (posB - tyB * OWt) : 0;

    for (int k0 = 0; k0 < Kpad; k0 += GEMM_TK) {
        // ---- stage A tile: 64 rows x 32 k, 16 halves per thread
        {
            const int mm = tid >> 1;
            const int kk = (tid & 1) << 4;
            const _Float16* src = A + (long)(mBase + mm) * Kpad + k0 + kk;
#pragma unroll
            for (int j = 0; j < 16; ++j) As[mm * LDA_H + kk + j] = src[j];
        }
        // ---- stage B tile: gather conv patch, f32 -> f16
        if (aligned) {
            // tap constant across chunk: hoist bounds check & base pointer
            const int tap = k0 / Cin;
            const int ci0 = k0 - tap * Cin;
            bool ok = validB;
            const float* base = nullptr;
            if (ok) {
                const int iy = tyB * stride + taps.dy[tap];
                const int ix = txB * stride + taps.dx[tap];
                ok = (iy >= 0) && (iy < IH) && (ix >= 0) && (ix < IW);
                if (ok) base = Xi + (long)ci0 * IHW + (long)iy * IW + ix;
            }
            const unsigned rowOff = (unsigned)(nB * LDB_H) >> 1;  // in dwords
            if (ok) {
#pragma unroll
                for (int g = 0; g < GEMM_TK; g += 8) {
                    float v[8];
#pragma unroll
                    for (int j = 0; j < 8; ++j) {
                        v[j] = base[(long)(g + j) * IHW];
                        if (relu_in) v[j] = fmaxf(v[j], 0.0f);
                    }
#pragma unroll
                    for (int j = 0; j < 8; j += 2)
                        Bs32[rowOff + ((g + j) >> 1)] = pack_h2(v[j], v[j + 1]);
                }
            } else {
#pragma unroll
                for (int j = 0; j < GEMM_TK / 2; ++j) Bs32[rowOff + j] = 0u;
            }
        } else {
            // generic path (only ec1: Cin=3, Kpad>Ktot)
            int tap = k0 / Cin;
            int ci  = k0 - tap * Cin;
#pragma unroll 4
            for (int kk = 0; kk < GEMM_TK; ++kk) {
                const int k = k0 + kk;
                float v = 0.0f;
                if (validB && k < Ktot) {
                    const int iy = tyB * stride + taps.dy[tap];
                    const int ix = txB * stride + taps.dx[tap];
                    if (iy >= 0 && iy < IH && ix >= 0 && ix < IW) {
                        v = Xi[ci * IHW + (long)iy * IW + ix];
                        if (relu_in) v = fmaxf(v, 0.0f);
                    }
                }
                Bs[nB * LDB_H + kk] = (_Float16)v;
                if (++ci == Cin) { ci = 0; ++tap; }
            }
        }
        __syncthreads();

        // ---- fragments + WMMA
        v16h af[2], bf[4];
#pragma unroll
        for (int ms = 0; ms < 2; ++ms) {
            const int m = wm + ms * 16 + l16;
#pragma unroll
            for (int j = 0; j < 16; ++j) {
                const int kq = j + 16 * half_;  // contiguous k = 16*half + j (same as B)
                af[ms][j] = As[m * LDA_H + kq];
            }
        }
#pragma unroll
        for (int ns = 0; ns < 4; ++ns) {
            const int n  = wn + ns * 16 + l16;
            const int kb = half_ << 4;                            // contiguous k = 16*half + j (same as A)
#pragma unroll
            for (int j = 0; j < 16; ++j) bf[ns][j] = Bs[n * LDB_H + kb + j];
        }
#pragma unroll
        for (int ms = 0; ms < 2; ++ms)
#pragma unroll
            for (int ns = 0; ns < 4; ++ns)
                acc[ms][ns] = __builtin_amdgcn_wmma_f32_16x16x32_f16(
                    false, af[ms], false, bf[ns], (short)0, acc[ms][ns], false, false);
        __syncthreads();
    }

    // ---- epilogue: C layout lane(N)=lane&15, row m = r + 8*half
#pragma unroll
    for (int ns = 0; ns < 4; ++ns) {
        const int pos = posBase + wn + ns * 16 + l16;
        if (pos >= Npos) continue;
        const int ty = pos / OWt;
        const int tx = pos - ty * OWt;
        const int oy = oy0 + ty * oys;
        const int ox = ox0 + tx * oxs;
#pragma unroll
        for (int ms = 0; ms < 2; ++ms) {
#pragma unroll
            for (int r = 0; r < 8; ++r) {
                const int co = mBase + wm + ms * 16 + half_ * 8 + r;
                Y[(((long)img * Cout + co) * OHf + oy) * OWf + ox] =
                    acc[ms][ns][r] + bias[co];
            }
        }
    }
}

// ---------------------------------------------------------------------------
// Pack f32 conv weights into f16 A matrix [Cout, Kpad], k = tap*Cin + ci.
// ---------------------------------------------------------------------------
__global__ void pack_w(const float* __restrict__ W, _Float16* __restrict__ A,
                       int Cout, int Cin, int Ktot, int Kpad,
                       int s_co, int s_ci, KMap km)
{
    const long total = (long)Cout * Kpad;
    for (long i = blockIdx.x * 256L + threadIdx.x; i < total; i += (long)gridDim.x * 256) {
        const int co = (int)(i / Kpad);
        const int k  = (int)(i - (long)co * Kpad);
        _Float16 v = (_Float16)0.0f;
        if (k < Ktot) {
            const int tap = k / Cin;
            const int ci  = k - tap * Cin;
            v = (_Float16)W[(long)co * s_co + (long)ci * s_ci + km.idx[tap]];
        }
        A[i] = v;
    }
}

// ---------------------------------------------------------------------------
// BatchNorm (training mode) — deterministic two-stage reduction.
// One block per (channel, image) plane: no per-element divisions.
// ---------------------------------------------------------------------------
__global__ void bn_reduce(const float* __restrict__ X, float* __restrict__ part,
                          int C, int HW)
{
    const int c = blockIdx.x, n = blockIdx.y, tid = threadIdx.x;
    const float* p = X + ((long)n * C + c) * HW;
    float s0 = 0.f, s1 = 0.f;
    for (int i = tid; i < HW; i += 256) {
        const float v = p[i];
        s0 += v; s1 += v * v;
    }
    __shared__ float r0[256], r1[256];
    r0[tid] = s0; r1[tid] = s1;
    __syncthreads();
    for (int o = 128; o > 0; o >>= 1) {
        if (tid < o) { r0[tid] += r0[tid + o]; r1[tid] += r1[tid + o]; }
        __syncthreads();
    }
    if (tid == 0) {
        part[((long)c * 32 + n) * 2 + 0] = r0[0];
        part[((long)c * 32 + n) * 2 + 1] = r1[0];
    }
}

__global__ void bn_final(const float* __restrict__ part, float* __restrict__ stats,
                         int C, float invCnt)
{
    const int c = blockIdx.x * blockDim.x + threadIdx.x;
    if (c >= C) return;
    float s0 = 0.f, s1 = 0.f;
    for (int s = 0; s < 32; ++s) {
        s0 += part[((long)c * 32 + s) * 2 + 0];
        s1 += part[((long)c * 32 + s) * 2 + 1];
    }
    const float m = s0 * invCnt;
    const float var = s1 * invCnt - m * m;
    stats[2 * c]     = m;
    stats[2 * c + 1] = rsqrtf(var + 1e-5f);
}

// grid = (ceil(HW/256), C, N): channel constants become scalar loads.
__global__ void bn_apply(const float* __restrict__ X, float* __restrict__ Y,
                         const float* __restrict__ stats, const float* __restrict__ g,
                         const float* __restrict__ b, const float* __restrict__ res,
                         int HW, int C, int relu)
{
    const int c = blockIdx.y, n = blockIdx.z;
    const int i = blockIdx.x * 256 + threadIdx.x;
    if (i >= HW) return;
    const long base = ((long)n * C + c) * HW + i;
    const float mu = stats[2 * c], inv = stats[2 * c + 1];
    float y = (X[base] - mu) * inv * g[c] + b[c];
    if (relu) y = fmaxf(y, 0.f);
    if (res)  y += res[base];
    Y[base] = y;
}

__global__ void fill_bias(float* __restrict__ Y, const float* __restrict__ bias,
                          int HW, int C)
{
    const int c = blockIdx.y, n = blockIdx.z;
    const int i = blockIdx.x * 256 + threadIdx.x;
    if (i >= HW) return;
    Y[((long)n * C + c) * HW + i] = bias[c];
}

// ---------------------------------------------------------------------------
// VQ nearest-neighbor: codebook (512x64) + |e|^2 in LDS (132KB of 320KB WGP LDS)
// ---------------------------------------------------------------------------
__global__ void vq_nn(const float* __restrict__ ze, const float* __restrict__ cb,
                      float* __restrict__ zq, int HW, int CHW, int total)
{
    __shared__ float E[512 * 64];
    __shared__ float e2[512];
    for (int i = threadIdx.x; i < 512 * 64; i += 256) E[i] = cb[i];
    __syncthreads();
    for (int k = threadIdx.x; k < 512; k += 256) {
        float s = 0.f;
        for (int d = 0; d < 64; ++d) { const float e = E[k * 64 + d]; s += e * e; }
        e2[k] = s;
    }
    __syncthreads();
    const int p = blockIdx.x * 256 + threadIdx.x;
    if (p >= total) return;
    const int n = p / HW;
    const int r = p - n * HW;
    const float* zp = ze + (long)n * CHW + r;
    float z[64];
#pragma unroll
    for (int d = 0; d < 64; ++d) z[d] = zp[(long)d * HW];
    int best = 0; float bd = 3.4e38f;
    for (int k = 0; k < 512; ++k) {
        float dot = 0.f;
        const float* ek = &E[k * 64];
#pragma unroll 8
        for (int d = 0; d < 64; ++d) dot += z[d] * ek[d];
        const float s = e2[k] - 2.f * dot;   // argmin invariant to +|z|^2
        if (s < bd) { bd = s; best = k; }    // strict < == first-min (jnp.argmin)
    }
    float* qo = zq + (long)n * CHW + r;
    const float* eb = &E[best * 64];
#pragma unroll
    for (int d = 0; d < 64; ++d) qo[(long)d * HW] = eb[d];
}

// ---------------------------------------------------------------------------
// Final ConvT 128->3 (k4,s2,p1) + tanh. grid = (ox-blocks, OH, B): no divides.
// ---------------------------------------------------------------------------
__global__ void convT_final_tanh(const float* __restrict__ X, const float* __restrict__ W,
                                 const float* __restrict__ bias, float* __restrict__ Y,
                                 int IH, int IW, int OH, int OW)
{
    __shared__ float Wl[128 * 48];   // [ci][co][ky*4+kx]
    for (int i = threadIdx.x; i < 128 * 48; i += 256) Wl[i] = W[i];
    __syncthreads();
    const int ox = blockIdx.x * 256 + threadIdx.x;
    const int oy = blockIdx.y;
    const int n  = blockIdx.z;
    if (ox >= OW) return;
    const long IHW = (long)IH * IW;
    const long OHW = (long)OH * OW;
    float a0 = 0.f, a1 = 0.f, a2 = 0.f;
    for (int ky = 0; ky < 4; ++ky) {
        const int t = oy + 1 - ky;
        if (t < 0 || (t & 1)) continue;
        const int iy = t >> 1; if (iy >= IH) continue;
        for (int kx = 0; kx < 4; ++kx) {
            const int u = ox + 1 - kx;
            if (u < 0 || (u & 1)) continue;
            const int ix = u >> 1; if (ix >= IW) continue;
            const float* xp = X + (long)n * 128 * IHW + (long)iy * IW + ix;
            const int wk = ky * 4 + kx;
            for (int ci = 0; ci < 128; ++ci) {
                const float v = xp[(long)ci * IHW];
                a0 = fmaf(v, Wl[ci * 48 +  0 + wk], a0);
                a1 = fmaf(v, Wl[ci * 48 + 16 + wk], a1);
                a2 = fmaf(v, Wl[ci * 48 + 32 + wk], a2);
            }
        }
    }
    float* yp = Y + (long)n * 3 * OHW + (long)oy * OW + ox;
    yp[0]       = tanhf(a0 + bias[0]);
    yp[OHW]     = tanhf(a1 + bias[1]);
    yp[2 * OHW] = tanhf(a2 + bias[2]);
}

// ===========================================================================
extern "C" void kernel_launch(void* const* d_in, const int* in_sizes, int n_in,
                              void* d_out, int out_size, void* d_ws, size_t ws_size,
                              hipStream_t stream)
{
    (void)in_sizes; (void)n_in; (void)out_size; (void)ws_size;
    const float* x      = (const float*)d_in[0];
    const float* ec1_w  = (const float*)d_in[1];
    const float* ec1_b  = (const float*)d_in[2];
    const float* ebn1_g = (const float*)d_in[3];
    const float* ebn1_b = (const float*)d_in[4];
    const float* ec2_w  = (const float*)d_in[5];
    const float* ec2_b  = (const float*)d_in[6];
    const float* eo_w   = (const float*)d_in[7];
    const float* eo_b   = (const float*)d_in[8];
    const float* eo_g   = (const float*)d_in[9];
    const float* eo_bb  = (const float*)d_in[10];
    const float* rp[4][8];      // er1, er2, dr1, dr2: w1,b1,g1,bb1,w2,b2,g2,bb2
    {
        const int bases[4] = {11, 19, 27, 35};
        for (int i = 0; i < 4; ++i)
            for (int j = 0; j < 8; ++j) rp[i][j] = (const float*)d_in[bases[i] + j];
    }
    const float* codebook = (const float*)d_in[43];
    const float* do_w  = (const float*)d_in[44];
    const float* do_b  = (const float*)d_in[45];
    const float* do_g  = (const float*)d_in[46];
    const float* do_bb = (const float*)d_in[47];
    const float* dc1_w = (const float*)d_in[48];
    const float* dc1_b = (const float*)d_in[49];
    const float* dbn1_g= (const float*)d_in[50];
    const float* dbn1_b= (const float*)d_in[51];
    const float* dc2_w = (const float*)d_in[52];
    const float* dc2_b = (const float*)d_in[53];

    // ---- outputs: x_tilde | z_e | z_q_bar (flat, in return order)
    float* out_x = (float*)d_out;
    float* ze = out_x + 32L * 3 * 272 * 272;
    float* zq = ze + 32L * 64 * 66 * 66;

    // ---- workspace partition (t1 aliases t9, t2 aliases t8)
    char* ws = (char*)d_ws;
    size_t off = 0;
    auto alloc = [&](size_t bytes) -> char* {
        char* p = ws + off;
        off += (bytes + 255) & ~(size_t)255;
        return p;
    };
    float* regA = (float*)alloc(302841856);   // t1 (268MB) then t9 (303MB)
    float* t1 = regA, *t9 = regA;
    float* regB = (float*)alloc(151519232);   // t2 (134MB) then t8 (152MB)
    float* t2 = regB, *t8 = regB;
    float* rA = (float*)alloc(35684352);
    float* rB = (float*)alloc(35684352);
    float* rC = (float*)alloc(35684352);
    _Float16* wEc1 = (_Float16*)alloc(128 * 64 * 2);
    _Float16* wEc2 = (_Float16*)alloc(256 * 2048 * 2);
    _Float16* wEo  = (_Float16*)alloc(64 * 256 * 2);
    _Float16* wR1[4], *wR2[4];
    for (int i = 0; i < 4; ++i) { wR1[i] = (_Float16*)alloc(64 * 576 * 2); wR2[i] = (_Float16*)alloc(64 * 64 * 2); }
    _Float16* wDo = (_Float16*)alloc(256 * 64 * 2);
    _Float16* wDc1[4];
    for (int i = 0; i < 4; ++i) wDc1[i] = (_Float16*)alloc(128 * 1024 * 2);
    float* part  = (float*)alloc(256 * 32 * 2 * 4);
    float* stats = (float*)alloc(2048);

    // ---- tap tables
    Taps tap44{}; KMap km16{};
    tap44.n = 16;
    for (int t = 0; t < 16; ++t) { tap44.dy[t] = t / 4 - 1; tap44.dx[t] = t % 4 - 1; km16.idx[t] = t; }
    Taps tap33{}; KMap km9{};
    tap33.n = 9;
    for (int t = 0; t < 9; ++t) { tap33.dy[t] = t / 3 - 1; tap33.dx[t] = t % 3 - 1; km9.idx[t] = t; }
    Taps tap11{}; KMap km1{};
    tap11.n = 1; tap11.dy[0] = 0; tap11.dx[0] = 0; km1.idx[0] = 0;
    Taps tapT[4]; KMap kmT[4];
    for (int py = 0; py < 2; ++py)
        for (int px = 0; px < 2; ++px) {
            const int c = py * 2 + px;
            tapT[c] = Taps{}; kmT[c] = KMap{};
            int kyl[2], dyl[2], kxl[2], dxl[2];
            if (py == 0) { kyl[0] = 1; dyl[0] = 0; kyl[1] = 3; dyl[1] = -1; }
            else         { kyl[0] = 0; dyl[0] = 1; kyl[1] = 2; dyl[1] =  0; }
            if (px == 0) { kxl[0] = 1; dxl[0] = 0; kxl[1] = 3; dxl[1] = -1; }
            else         { kxl[0] = 0; dxl[0] = 1; kxl[1] = 2; dxl[1] =  0; }
            tapT[c].n = 4;
            for (int a = 0; a < 2; ++a)
                for (int b = 0; b < 2; ++b) {
                    const int t = a * 2 + b;
                    tapT[c].dy[t] = dyl[a]; tapT[c].dx[t] = dxl[b];
                    kmT[c].idx[t] = kyl[a] * 4 + kxl[b];
                }
        }

    // ---- helpers
    auto pack = [&](const float* W, _Float16* A, int Cout, int Cin, int Ktot, int Kpad,
                    int s_co, int s_ci, const KMap& km) {
        const long total = (long)Cout * Kpad;
        pack_w<<<(int)((total + 255) / 256), 256, 0, stream>>>(W, A, Cout, Cin, Ktot, Kpad, s_co, s_ci, km);
    };
    auto gemm = [&](const _Float16* A, const float* Xb, const float* bias, float* Yb,
                    const Taps& taps, int Cin, int IH, int IW, int s, int relu,
                    int Ktot, int Kpad, int OHt, int OWt, int oy0, int ox0, int oys, int oxs,
                    int OHf, int OWf, int Cout) {
        dim3 grid((OHt * OWt + GEMM_BN - 1) / GEMM_BN, Cout / GEMM_BM, 32);
        conv_gemm_wmma<<<grid, 128, 0, stream>>>(A, Xb, bias, Yb, taps, Cin, IH, IW, s, relu,
                                                 Ktot, Kpad, OHt, OWt, oy0, ox0, oys, oxs,
                                                 OHf, OWf, Cout);
    };
    auto bn = [&](const float* Xb, float* Yb, const float* g, const float* b,
                  const float* res, int C, int HW, int relu) {
        bn_reduce<<<dim3(C, 32), 256, 0, stream>>>(Xb, part, C, HW);
        bn_final<<<(C + 63) / 64, 64, 0, stream>>>(part, stats, C, 1.0f / (32.0f * HW));
        bn_apply<<<dim3((HW + 255) / 256, C, 32), 256, 0, stream>>>(Xb, Yb, stats, g, b, res, HW, C, relu);
    };
    auto fillb = [&](float* Yb, const float* bias, int C, int HW) {
        fill_bias<<<dim3((HW + 255) / 256, C, 32), 256, 0, stream>>>(Yb, bias, HW, C);
    };

    // ---- pack all weights (f32 -> f16 GEMM-A layout)
    pack(ec1_w, wEc1, 128,   3,   48,   64,   48,   16, km16);
    pack(ec2_w, wEc2, 256, 128, 2048, 2048, 2048,   16, km16);
    pack(eo_w,  wEo,   64, 256,  256,  256,  256,    1, km1);
    for (int i = 0; i < 4; ++i) {
        pack(rp[i][0], wR1[i], 64, 64, 576, 576, 576, 9, km9);
        pack(rp[i][4], wR2[i], 64, 64,  64,  64,  64, 1, km1);
    }
    pack(do_w, wDo, 256, 64, 64, 64, 64, 1, km1);
    for (int c = 0; c < 4; ++c)
        pack(dc1_w, wDc1[c], 128, 256, 1024, 1024, 16, 2048, kmT[c]);  // [in,out,kh,kw]

    // ---- encoder
    gemm(wEc1, x, ec1_b, t1, tap44, 3, 256, 256, 2, 0, 48, 64,
         128, 128, 0, 0, 1, 1, 128, 128, 128);
    bn(t1, t1, ebn1_g, ebn1_b, nullptr, 128, 128 * 128, 0);            // LeakyReLU(1.0)=identity
    gemm(wEc2, t1, ec2_b, t2, tap44, 128, 128, 128, 2, 0, 2048, 2048,
         64, 64, 0, 0, 1, 1, 64, 64, 256);
    // one_by_one: relu -> 1x1 conv (pad=1 => 66x66, border = bias) -> BN -> relu
    fillb(rA, eo_b, 64, 66 * 66);
    gemm(wEo, t2, eo_b, rA, tap11, 256, 64, 64, 1, 1, 256, 256,
         64, 64, 1, 1, 1, 1, 66, 66, 64);
    bn(rA, rA, eo_g, eo_bb, nullptr, 64, 66 * 66, 1);

    auto resblk = [&](const float* in, float* ta, float* tb, float* out, int i) {
        gemm(wR1[i], in, rp[i][1], ta, tap33, 64, 66, 66, 1, 0, 576, 576,
             66, 66, 0, 0, 1, 1, 66, 66, 64);
        bn(ta, ta, rp[i][2], rp[i][3], nullptr, 64, 66 * 66, 1);
        gemm(wR2[i], ta, rp[i][5], tb, tap11, 64, 66, 66, 1, 0, 64, 64,
             66, 66, 0, 0, 1, 1, 66, 66, 64);
        bn(tb, out, rp[i][6], rp[i][7], in, 64, 66 * 66, 0);           // out = in + h
    };
    resblk(rA, rB, rC, rB, 0);     // er1
    resblk(rB, rC, rA, ze, 1);     // er2 -> z_e (written straight into d_out)

    // ---- VQ nearest neighbor -> z_q_bar (into d_out)
    vq_nn<<<(32 * 66 * 66 + 255) / 256, 256, 0, stream>>>(
        ze, codebook, zq, 66 * 66, 64 * 66 * 66, 32 * 66 * 66);

    // ---- decoder (z_q_st == z_q_bar numerically in forward)
    resblk(zq, rA, rB, rA, 2);     // dr1
    resblk(rA, rB, rC, rB, 3);     // dr2
    fillb(t8, do_b, 256, 68 * 68);
    gemm(wDo, rB, do_b, t8, tap11, 64, 66, 66, 1, 1, 64, 64,
         66, 66, 1, 1, 1, 1, 68, 68, 256);
    bn(t8, t8, do_g, do_bb, nullptr, 256, 68 * 68, 1);
    // ConvT 256->128 k4 s2 p1 as 4 output-parity GEMMs (K = 256*4 taps)
    for (int py = 0; py < 2; ++py)
        for (int px = 0; px < 2; ++px) {
            const int c = py * 2 + px;
            gemm(wDc1[c], t8, dc1_b, t9, tapT[c], 256, 68, 68, 1, 0, 1024, 1024,
                 68, 68, py, px, 2, 2, 136, 136, 128);
        }
    bn(t9, t9, dbn1_g, dbn1_b, nullptr, 128, 136 * 136, 0);
    // ConvT 128->3 + tanh -> x_tilde
    convT_final_tanh<<<dim3((272 + 255) / 256, 272, 32), 256, 0, stream>>>(
        t9, dc2_w, dc2_b, out_x, 136, 136, 272, 272);
}
